// TransformerGateNetwork_87265145520221
// MI455X (gfx1250) — matrix operand order repaired and checked by reading, not compiled, not run on hardware
//
#include <hip/hip_runtime.h>
#include <stdint.h>

#define H    512
#define NH   8
#define NE   131072
#define NBLK 256
#define CHUNK 512   // NE / NBLK

typedef __attribute__((ext_vector_type(16))) __bf16 v16bf;
typedef __attribute__((ext_vector_type(2)))  __bf16 v2bf;
typedef __attribute__((ext_vector_type(8)))  float  v8f;

union BF16Frag { uint32_t u[8]; uint4 q[2]; v16bf v; };

__device__ __forceinline__ uint16_t f2bf(float f) {
  uint32_t u = __float_as_uint(f);
  u += 0x7FFFu + ((u >> 16) & 1u);   // round-to-nearest-even
  return (uint16_t)(u >> 16);
}
__device__ __forceinline__ uint32_t pack2bf(float a, float b) {
#if __has_builtin(__builtin_amdgcn_cvt_pk_bf16_f32)
  union { v2bf v; uint32_t u; } cv;
  cv.v = __builtin_amdgcn_cvt_pk_bf16_f32(a, b);
  return cv.u;
#else
  return (uint32_t)f2bf(a) | ((uint32_t)f2bf(b) << 16);
#endif
}
__device__ __forceinline__ v8f wmma_bf16(const BF16Frag& a, const BF16Frag& b, v8f c) {
  return __builtin_amdgcn_wmma_f32_16x16x32_bf16(false, a.v, false, b.v, (short)0, c, false, false);
}

// ---------------- pack efp_w into per-lane-contiguous B-fragment order ----------------
// wB[((nt*16+kt)*32 + lane)*8 + v] = pair (W[n][K], W[n][K+1]),
//   n = nt*16 + lane%16, K = kt*32 + (v<4 ? 8*(lane/16) + 2v : 16 + 8*(lane/16) + 2(v-4))   -> lane loads 2x b128
__global__ void k_pack_w(const float* __restrict__ W, uint32_t* __restrict__ wB) {
  int kt = blockIdx.x, nt = blockIdx.y;
  int t = threadIdx.x;
  int lane = t >> 3, v = t & 7;
  int n = nt * 16 + (lane & 15);
  int K = kt * 32 + ((v < 4) ? (8 * (lane >> 4) + 2 * v) : (16 + 8 * (lane >> 4) + 2 * (v - 4)));
  wB[(size_t)(nt * 16 + kt) * 256 + t] = pack2bf(W[n * H + K], W[n * H + K + 1]);
}

// ---------------- query vector: q = LN(relu(pair @ npp_w^T + b)) ----------------
__global__ void k_qnode(const float* __restrict__ ni, const float* __restrict__ nj,
                        const float* __restrict__ w, const float* __restrict__ b,
                        const float* __restrict__ g, const float* __restrict__ beta,
                        float* __restrict__ qout) {
  __shared__ float s1[512], s2[512];
  int n = threadIdx.x;
  float acc = b[n];
  const float* row = w + (size_t)n * 1024;
  for (int k = 0; k < 512; ++k) acc += ni[k] * row[k];
  for (int k = 0; k < 512; ++k) acc += nj[k] * row[512 + k];
  float y = fmaxf(acc, 0.f);
  s1[n] = y; s2[n] = y * y;
  __syncthreads();
  for (int off = 256; off > 0; off >>= 1) {
    if (n < off) { s1[n] += s1[n + off]; s2[n] += s2[n + off]; }
    __syncthreads();
  }
  float mean = s1[0] * (1.f / 512.f);
  float var  = s2[0] * (1.f / 512.f) - mean * mean;
  qout[n] = (y - mean) * rsqrtf(var + 1e-5f) * g[n] + beta[n];
}

// ---------------- edge feature projection (big WMMA GEMM) ----------------
__global__ void __launch_bounds__(256) k_efp(
    const float* __restrict__ ef, const int* __restrict__ et,
    const float* __restrict__ temb, const uint32_t* __restrict__ wB,
    const float* __restrict__ bias, const float* __restrict__ g,
    const float* __restrict__ beta, unsigned short* __restrict__ xbf) {
  __shared__ float yt[16 * 512];
  int wave = threadIdx.x >> 5, lane = threadIdx.x & 31;
  int half = lane >> 4;
  int row = blockIdx.x * 16 + (lane & 15);
  int ty = et[row];
  const float* efr = ef + (size_t)row * H;
  const float* emr = temb + (size_t)ty * H;
  v8f C[4] = {};
  int ntbase = wave * 4;
  for (int kt = 0; kt < 16; ++kt) {
    // A: consecutive-f32 groups: K = kt*32 + 8*half + [0..7] and + 16
    int K0 = kt * 32 + 8 * half;
    float4 ea = *(const float4*)(efr + K0);
    float4 eb = *(const float4*)(efr + K0 + 4);
    float4 ec = *(const float4*)(efr + K0 + 16);
    float4 ed = *(const float4*)(efr + K0 + 20);
    float4 ma = *(const float4*)(emr + K0);
    float4 mb = *(const float4*)(emr + K0 + 4);
    float4 mc = *(const float4*)(emr + K0 + 16);
    float4 md = *(const float4*)(emr + K0 + 20);
    BF16Frag a;
    a.u[0] = pack2bf(ea.x + ma.x, ea.y + ma.y);
    a.u[1] = pack2bf(ea.z + ma.z, ea.w + ma.w);
    a.u[2] = pack2bf(eb.x + mb.x, eb.y + mb.y);
    a.u[3] = pack2bf(eb.z + mb.z, eb.w + mb.w);
    a.u[4] = pack2bf(ec.x + mc.x, ec.y + mc.y);
    a.u[5] = pack2bf(ec.z + mc.z, ec.w + mc.w);
    a.u[6] = pack2bf(ed.x + md.x, ed.y + md.y);
    a.u[7] = pack2bf(ed.z + md.z, ed.w + md.w);
#pragma unroll
    for (int j = 0; j < 4; ++j) {
      BF16Frag b;
      const uint4* bp = (const uint4*)(wB + (size_t)((ntbase + j) * 16 + kt) * 256 + lane * 8);
      b.q[0] = bp[0];
      b.q[1] = bp[1];
      C[j] = wmma_bf16(a, b, C[j]);
    }
  }
  // bias + relu into LDS (C layout: lane -> col N, vgpr r -> row M = r + 8*half)
#pragma unroll
  for (int j = 0; j < 4; ++j) {
    int col = (ntbase + j) * 16 + (lane & 15);
    float bc = bias[col];
#pragma unroll
    for (int r = 0; r < 8; ++r)
      yt[(r + 8 * half) * 512 + col] = fmaxf(C[j][r] + bc, 0.f);
  }
  __syncthreads();
  // layernorm per row; wave handles rows {wave, wave+8}
  for (int rr = 0; rr < 2; ++rr) {
    int M = wave + rr * 8;
    float s = 0.f, s2 = 0.f;
    for (int i = 0; i < 16; ++i) {
      float yv = yt[M * 512 + lane + i * 32];
      s += yv; s2 += yv * yv;
    }
#pragma unroll
    for (int off = 16; off > 0; off >>= 1) {
      s += __shfl_xor(s, off);
      s2 += __shfl_xor(s2, off);
    }
    float mean = s * (1.f / 512.f);
    float rs = rsqrtf(s2 * (1.f / 512.f) - mean * mean + 1e-5f);
    size_t ob = (size_t)(blockIdx.x * 16 + M) * H;
    for (int i = 0; i < 16; ++i) {
      int col = lane + i * 32;
      xbf[ob + col] = f2bf((yt[M * 512 + col] - mean) * rs * g[col] + beta[col]);
    }
  }
}

// ---------------- per-layer prep: qh, folded wkq (packed bf16 B-frags), cq ----------------
__global__ void k_layer_prep(const float* __restrict__ qin, const float* __restrict__ w_in,
                             const float* __restrict__ b_in, uint32_t* __restrict__ wkqB,
                             float* __restrict__ cq) {
  __shared__ float qh[512];
  __shared__ float wkq[NH * 512];
  int t = threadIdx.x;
  {
    float acc = b_in[t];
    const float* r = w_in + (size_t)t * 512;  // wq rows 0..511
    for (int k = 0; k < 512; ++k) acc += qin[k] * r[k];
    qh[t] = acc;
  }
  __syncthreads();
  for (int i = 0; i < 8; ++i) {
    int idx = t + i * 512;
    int h = idx >> 9, k = idx & 511;
    float acc = 0.f;
    const float* wkb = w_in + (size_t)(512 + h * 64) * 512 + k;  // wk rows 512..1023
    const float* qb = qh + h * 64;
    for (int d = 0; d < 64; ++d) acc += qb[d] * wkb[(size_t)d * 512];
    wkq[idx] = acc;
  }
  __syncthreads();
  for (int i = 0; i < 8; ++i) {
    int idx = t + i * 512;               // = kt*256 + lane*8 + v (per-lane contiguous)
    int kt = idx >> 8, lane = (idx >> 3) & 31, v = idx & 7;
    int n = lane & 15;
    int K = kt * 32 + ((v < 4) ? (8 * (lane >> 4) + 2 * v) : (16 + 8 * (lane >> 4) + 2 * (v - 4)));
    float a = (n < NH) ? wkq[n * 512 + K] : 0.f;
    float b = (n < NH) ? wkq[n * 512 + K + 1] : 0.f;
    wkqB[idx] = pack2bf(a, b);
  }
  if (t < NH) {
    float acc = 0.f;
    for (int d = 0; d < 64; ++d) acc += qh[t * 64 + d] * b_in[512 + t * 64 + d];
    cq[t] = acc;
  }
}

// ---------------- attention main: scores + block-local softmax + z partials ----------------
__global__ void __launch_bounds__(256) k_attn_main(
    const unsigned short* __restrict__ xbf, const uint32_t* __restrict__ wkqB,
    const float* __restrict__ cq, const float* __restrict__ pmask,
    float* __restrict__ scores, float* __restrict__ wsmax,
    float* __restrict__ wssum, float* __restrict__ zpart) {
  __shared__ __align__(16) unsigned short xT[512 * 32];
  __shared__ float wmaxs[8 * 8];
  __shared__ float bmax[8];
  __shared__ float bsum[8];
  const uint32_t* x32 = (const uint32_t*)xbf;
  int tid = threadIdx.x;
  int wave = tid >> 5, lane = tid & 31;
  int nlane = lane & 15, half = lane >> 4;
  int e0 = blockIdx.x * CHUNK;

  // ---- Phase A: scores = (x @ wkq^T + cq)/8, track block max ----
  float lmax = -3.0e38f;
  for (int mt = wave; mt < 32; mt += 8) {
    int rowbase = e0 + mt * 16;
    int row = rowbase + nlane;
    v8f C = {};
    for (int kt = 0; kt < 16; ++kt) {
      BF16Frag a, b;
      // A dwords are consecutive per lane-half: row*256 + kt*16 + 4*half (+8)
      const uint4* ap = (const uint4*)(x32 + (size_t)row * 256 + kt * 16 + 4 * half);
      a.q[0] = ap[0];
      a.q[1] = ap[2];
      const uint4* bp = (const uint4*)(wkqB + kt * 256 + lane * 8);
      b.q[0] = bp[0];
      b.q[1] = bp[1];
      C = wmma_bf16(a, b, C);
    }
    float cqn = (nlane < NH) ? cq[nlane] : 0.f;
#pragma unroll
    for (int r = 0; r < 8; ++r) {
      int e = rowbase + r + 8 * half;
      float s = (C[r] + cqn) * 0.125f;
      if (pmask[e] == 0.f) s = -1e30f;
      if (nlane < NH) {
        scores[(size_t)e * NH + nlane] = s;
        lmax = fmaxf(lmax, s);
      }
    }
  }
  lmax = fmaxf(lmax, __shfl_xor(lmax, 16));
  if (lane < NH) wmaxs[wave * 8 + lane] = lmax;
  __syncthreads();
  if (tid < NH) {
    float gm = -3.0e38f;
    for (int w2 = 0; w2 < 8; ++w2) gm = fmaxf(gm, wmaxs[w2 * 8 + tid]);
    bmax[tid] = gm;
    bsum[tid] = 0.f;
    wsmax[blockIdx.x * NH + tid] = gm;
  }
  __syncthreads();
  // ---- Phase A2: block-local exp sums (each thread owns one head) ----
  {
    int h = tid & 7;
    float gm = bmax[h];
    float ps = 0.f;
    for (int i = 0; i < 16; ++i) {
      int e = e0 + ((tid + i * 256) >> 3);
      ps += __expf(scores[(size_t)e * NH + h] - gm);
    }
    atomicAdd(&bsum[h], ps);
  }
  __syncthreads();
  if (tid < NH) wssum[blockIdx.x * NH + tid] = bsum[tid];

  // ---- Phase B: z = p @ x  (A = p[16h x 32e], B = xT tile [32e x 16cols]) ----
  v8f Z[4] = {};
  for (int kt2 = 0; kt2 < 16; ++kt2) {
    int ebase = e0 + kt2 * 32;
    __syncthreads();
    for (int i = 0; i < 32; ++i) {
      int didx = tid + i * 256;
      int r = didx >> 8, cp = didx & 255;
      uint32_t u = x32[(size_t)(ebase + r) * 256 + cp];
      xT[(2 * cp) * 32 + r] = (unsigned short)(u & 0xFFFFu);
      xT[(2 * cp + 1) * 32 + r] = (unsigned short)(u >> 16);
    }
    __syncthreads();
    BF16Frag a;
    float gm = (nlane < NH) ? bmax[nlane] : 0.f;
#pragma unroll
    for (int v = 0; v < 8; ++v) {
      int K = 16 * half + 2 * v;
      float p0 = 0.f, p1 = 0.f;
      if (nlane < NH) {
        p0 = __expf(scores[(size_t)(ebase + K) * NH + nlane] - gm);
        p1 = __expf(scores[(size_t)(ebase + K + 1) * NH + nlane] - gm);
      }
      a.u[v] = pack2bf(p0, p1);
    }
#pragma unroll
    for (int j = 0; j < 4; ++j) {
      int col = wave * 64 + j * 16 + nlane;
      BF16Frag b;
      const uint4* xp = (const uint4*)&xT[col * 32 + half * 16];  // 8 consecutive dwords
      b.q[0] = xp[0];
      b.q[1] = xp[1];
      Z[j] = wmma_bf16(a, b, Z[j]);
    }
  }
  if (half == 0) {  // rows M = r are heads 0..7
#pragma unroll
    for (int j = 0; j < 4; ++j) {
      int col = wave * 64 + j * 16 + nlane;
#pragma unroll
      for (int r = 0; r < 8; ++r)
        zpart[((size_t)blockIdx.x * NH + r) * 512 + col] = Z[j][r];
    }
  }
}

// ---------------- per-layer finish: global softmax combine + out projection ----------------
__global__ void __launch_bounds__(1024) k_layer_finish(
    const float* __restrict__ wsmax, const float* __restrict__ wssum,
    const float* __restrict__ zpart, const float* __restrict__ b_in,
    const float* __restrict__ w_in, const float* __restrict__ w_out,
    const float* __restrict__ b_out, float* __restrict__ outvec,
    float* __restrict__ gmax, float* __restrict__ denomg) {
  __shared__ float gm_s[8], dn_s[8];
  __shared__ float z_s[NH * 512];
  __shared__ float oat_s[512];
  int t = threadIdx.x;
  if (t < NH) {
    float gm = -3.0e38f;
    for (int b = 0; b < NBLK; ++b) gm = fmaxf(gm, wsmax[b * NH + t]);
    float dn = 0.f;
    for (int b = 0; b < NBLK; ++b) dn += wssum[b * NH + t] * __expf(wsmax[b * NH + t] - gm);
    gm_s[t] = gm; dn_s[t] = dn;
    gmax[t] = gm; denomg[t] = dn;
  }
  __syncthreads();
  for (int i = 0; i < 4; ++i) {
    int idx = t + i * 1024;
    int h = idx >> 9, k = idx & 511;
    float gm = gm_s[h];
    float acc = 0.f;
    for (int b = 0; b < NBLK; ++b)
      acc += zpart[((size_t)b * NH + h) * 512 + k] * __expf(wsmax[b * NH + h] - gm);
    z_s[idx] = acc / dn_s[h];
  }
  __syncthreads();
  if (t < 512) {
    int h = t >> 6;
    float acc = b_in[1024 + t];                          // bv
    const float* wvr = w_in + (size_t)(1024 + t) * 512;  // wv rows 1024..1535
    const float* zr = z_s + h * 512;
    for (int k = 0; k < 512; ++k) acc += zr[k] * wvr[k];
    oat_s[t] = acc;
  }
  __syncthreads();
  if (t < 512) {
    float acc = b_out[t];
    const float* wor = w_out + (size_t)t * 512;
    for (int k = 0; k < 512; ++k) acc += oat_s[k] * wor[k];
    outvec[t] = acc;
  }
}

// ---------------- final: edge_weights = mean_h softmax (last layer), masked ----------------
__global__ void k_final(const float* __restrict__ scores, const float* __restrict__ pmask,
                        const float* __restrict__ gmax, const float* __restrict__ denomg,
                        float* __restrict__ out) {
  int e = blockIdx.x * 256 + threadIdx.x;
  if (e >= NE) return;
  float w = 0.f;
  if (pmask[e] != 0.f) {
    float acc = 0.f;
#pragma unroll
    for (int h = 0; h < NH; ++h)
      acc += __expf(scores[(size_t)e * NH + h] - gmax[h]) / denomg[h];
    w = acc * 0.125f;
  }
  out[e] = w;
}

extern "C" void kernel_launch(void* const* d_in, const int* in_sizes, int n_in,
                              void* d_out, int out_size, void* d_ws, size_t ws_size,
                              hipStream_t stream) {
  const float* node_i = (const float*)d_in[0];
  const float* node_j = (const float*)d_in[1];
  const float* ef     = (const float*)d_in[2];
  const int*   etyp   = (const int*)d_in[3];
  const float* pmask  = (const float*)d_in[4];
  const float* temb   = (const float*)d_in[5];
  const float* efp_w  = (const float*)d_in[6];
  const float* efp_b  = (const float*)d_in[7];
  const float* efp_g  = (const float*)d_in[8];
  const float* efp_be = (const float*)d_in[9];
  const float* npp_w  = (const float*)d_in[10];
  const float* npp_b  = (const float*)d_in[11];
  const float* npp_g  = (const float*)d_in[12];
  const float* npp_be = (const float*)d_in[13];
  const float* w_in[2]  = {(const float*)d_in[14], (const float*)d_in[18]};
  const float* b_in[2]  = {(const float*)d_in[15], (const float*)d_in[19]};
  const float* w_out[2] = {(const float*)d_in[16], (const float*)d_in[20]};
  const float* b_out[2] = {(const float*)d_in[17], (const float*)d_in[21]};

  char* ws = (char*)d_ws;
  size_t off = 0;
  auto alloc = [&](size_t bytes) -> void* {
    void* p = ws + off;
    off = (off + bytes + 255) & ~(size_t)255;
    return p;
  };
  unsigned short* xbf = (unsigned short*)alloc((size_t)NE * H * 2);  // 128 MB, L2-resident
  uint32_t* wB    = (uint32_t*)alloc((size_t)32 * 16 * 256 * 4);     // 512 KB
  float* q0       = (float*)alloc(H * 4);
  float* out0     = (float*)alloc(H * 4);
  uint32_t* wkqB  = (uint32_t*)alloc((size_t)16 * 256 * 4);
  float* cq       = (float*)alloc(64);
  float* scores   = (float*)alloc((size_t)NE * NH * 4);              // 4 MB
  float* wsmax    = (float*)alloc((size_t)NBLK * NH * 4);
  float* wssum    = (float*)alloc((size_t)NBLK * NH * 4);
  float* zpart    = (float*)alloc((size_t)NBLK * NH * 512 * 4);      // 4 MB
  float* gmaxb    = (float*)alloc(64);
  float* denomg   = (float*)alloc(64);

  k_pack_w<<<dim3(16, 32), 256, 0, stream>>>(efp_w, wB);
  k_qnode<<<1, 512, 0, stream>>>(node_i, node_j, npp_w, npp_b, npp_g, npp_be, q0);
  k_efp<<<NE / 16, 256, 0, stream>>>(ef, etyp, temb, wB, efp_b, efp_g, efp_be, xbf);
  for (int l = 0; l < 2; ++l) {
    const float* qin = (l == 0) ? q0 : out0;
    k_layer_prep<<<1, 512, 0, stream>>>(qin, w_in[l], b_in[l], wkqB, cq);
    k_attn_main<<<NBLK, 256, 0, stream>>>(xbf, wkqB, cq, pmask, scores, wsmax, wssum, zpart);
    k_layer_finish<<<1, 1024, 0, stream>>>(wsmax, wssum, zpart, b_in[l], w_in[l],
                                           w_out[l], b_out[l], out0, gmaxb, denomg);
  }
  k_final<<<NE / 256, 256, 0, stream>>>(scores, pmask, gmaxb, denomg, (float*)d_out);
}
